// GroupQueryAttention_61710090109333
// MI455X (gfx1250) — compile-verified, matrix operand order checked by reading, not run on hardware
//
#include <hip/hip_runtime.h>
#include <hip/hip_bf16.h>
#include <cstdint>

// ---------------------------------------------------------------------------
// GQA attention block for MI455X (gfx1250, wave32, WMMA).
// Shapes: B=1, N=2048, DIM=2048, H=32, KVH=4, D=64, REP=8.
// GEMMs: C = A * B^T, bf16 in / f32 accum, 128x128 block tiles double-buffered
// through LDS (async global->LDS, ASYNCcnt protocol), 8 waves x (32x64 strip)
// = 8 wmma per 32-K chunk per wave.
// Attention: flash-style, one wave per (head, 16 query rows), online softmax.
// ---------------------------------------------------------------------------

typedef __attribute__((ext_vector_type(16))) __bf16 bfx16;
typedef __attribute__((ext_vector_type(8)))  __bf16 bfx8;
typedef __attribute__((ext_vector_type(8)))  float  fx8;
typedef __attribute__((ext_vector_type(4)))  int    i32x4;

#define NTOK   2048
#define DMODEL 2048
#define NHQ    32
#define NHKV   4
#define DH     64
#define KVD    (NHKV * DH)   // 256
#define LDSS   40            // padded LDS row stride (elems): 16B aligned, conflict-free b128
#define PLDS_STRIDE 40

// ---- CDNA5 async global->LDS staging (guarded; fallback = load + ds_store) ----
#if defined(__has_builtin)
#  if __has_builtin(__builtin_amdgcn_global_load_async_to_lds_b128)
#    define USE_ASYNC_LDS 1
#  endif
#endif
#ifndef USE_ASYNC_LDS
#  define USE_ASYNC_LDS 0
#endif

#if USE_ASYNC_LDS
#  if __has_builtin(__builtin_amdgcn_s_wait_asynccnt)
#    define WAIT_ASYNC(n) __builtin_amdgcn_s_wait_asynccnt(n)
#  else
#    define WAIT_ASYNC(n) asm volatile("s_wait_asynccnt %0" ::"i"(n) : "memory")
#  endif
#else
#  define WAIT_ASYNC(n) ((void)0)
#endif

static __device__ __forceinline__ void stage_chunk(const __bf16* gp, __bf16* lp) {
#if USE_ASYNC_LDS
  // Signature (from clang diagnostic): arg0 = global i32x4* (AS1, non-const),
  // arg1 = LDS i32x4* (AS3), then imm offset, imm cpol.
  __builtin_amdgcn_global_load_async_to_lds_b128(
      (__attribute__((address_space(1))) i32x4*)(void*)gp,
      (__attribute__((address_space(3))) i32x4*)(void*)lp,
      0, 0);
#else
  *(bfx8*)lp = *(const bfx8*)gp;
#endif
}

// Load one 16x32 bf16 WMMA fragment from a row-major matrix (A layout; the
// B layout for X*W^T is identical). Lane l: row = row0 + (l&15); two b128
// chunks at k0 + (l>=16?8:0) and +16 more.
static __device__ __forceinline__ bfx16 load_frag(const __bf16* __restrict__ base,
                                                  int ld, int row0, int k0) {
  const int lane = threadIdx.x & 31;
  const __bf16* p = base + (size_t)(row0 + (lane & 15)) * ld + k0 + ((lane >> 4) << 3);
  bfx8 lo = *(const bfx8*)p;
  bfx8 hi = *(const bfx8*)(p + 16);
  bfx16 r;
#pragma unroll
  for (int i = 0; i < 8; ++i) { r[i] = lo[i]; r[i + 8] = hi[i]; }
  return r;
}

// Same fragment gather from an LDS tile with row stride LDSS.
static __device__ __forceinline__ bfx16 load_frag_lds(const __bf16* s, int row0) {
  const int lane = threadIdx.x & 31;
  const __bf16* p = s + (row0 + (lane & 15)) * LDSS + ((lane >> 4) << 3);
  bfx8 lo = *(const bfx8*)p;
  bfx8 hi = *(const bfx8*)(p + 16);
  bfx16 r;
#pragma unroll
  for (int i = 0; i < 8; ++i) { r[i] = lo[i]; r[i + 8] = hi[i]; }
  return r;
}

static __device__ __forceinline__ void store_val(float*  p, float v) { *p = v; }
static __device__ __forceinline__ void store_val(__bf16* p, float v) { *p = (__bf16)v; }

// ---------------------------------------------------------------------------
// C[M,N] = A[M,K] * B[N,K]^T.  Block = 256 thr = 8 waves, tile 128x128.
// Wave (wm = wave>>1, wn = wave&1) computes rows [wm*32,+32) x cols [wn*64,+64)
// => 2x4 accumulators, 8 wmma per 32-K chunk. A/B K-chunks are double-buffered
// in LDS; staging uses async global->LDS (ASYNCcnt) when available.
// ---------------------------------------------------------------------------
template <typename OutT>
__global__ __launch_bounds__(256) void gemm_abt_kernel(const __bf16* __restrict__ A,
                                                       const __bf16* __restrict__ B,
                                                       OutT* __restrict__ C,
                                                       int M, int N, int K) {
  __shared__ alignas(16) __bf16 sA[2][128 * LDSS];  // 10240 B each
  __shared__ alignas(16) __bf16 sB[2][128 * LDSS];  // total 40 KB
  const int tid   = threadIdx.x;
  const int wave  = tid >> 5;
  const int lane  = tid & 31;
  const int coln  = lane & 15;
  const int half8 = (lane >> 4) << 3;
  const int wm    = wave >> 1;   // 0..3
  const int wn    = wave & 1;    // 0..1
  const int m0    = blockIdx.x * 128;
  const int n0    = blockIdx.y * 128;

  fx8 acc[2][4] = {};

  // Stage one 128x32 K-chunk of A and of B into LDS buffer `buf`.
  // 512 x 16B chunks per tile / 256 threads = 2 chunks each per tile.
  auto stage = [&](int buf, int k0) {
#pragma unroll
    for (int r = 0; r < 2; ++r) {
      const int c   = tid + (r << 8);
      const int row = c >> 2;
      const int kc  = (c & 3) << 3;
      stage_chunk(A + (size_t)(m0 + row) * K + k0 + kc, &sA[buf][row * LDSS + kc]);
      stage_chunk(B + (size_t)(n0 + row) * K + k0 + kc, &sB[buf][row * LDSS + kc]);
    }
  };

  stage(0, 0);                       // 4 async ops in flight per thread
  const int nk = K >> 5;
  for (int ki = 0; ki < nk; ++ki) {
    const int cur = ki & 1;
    if (ki + 1 < nk) {
      stage(cur ^ 1, (ki + 1) << 5); // prefetch next chunk (8 in flight)
      WAIT_ASYNC(4);                 // oldest 4 (current chunk) complete
    } else {
      WAIT_ASYNC(0);
    }
    __syncthreads();                 // everyone's current-chunk copies visible

    bfx16 af[2];
#pragma unroll
    for (int rb = 0; rb < 2; ++rb)
      af[rb] = load_frag_lds(sA[cur], wm * 32 + rb * 16);
#pragma unroll
    for (int c = 0; c < 4; ++c) {
      bfx16 bf = load_frag_lds(sB[cur], wn * 64 + c * 16);
#pragma unroll
      for (int rb = 0; rb < 2; ++rb)
        acc[rb][c] = __builtin_amdgcn_wmma_f32_16x16x32_bf16(
            false, af[rb], false, bf, (short)0, acc[rb][c], false, false);
    }
    __syncthreads();                 // reads done before next-next overwrite
  }

  // C/D layout: lane holds one column; VGPR j holds row (+half8).
#pragma unroll
  for (int rb = 0; rb < 2; ++rb)
#pragma unroll
    for (int c = 0; c < 4; ++c) {
      const int n    = n0 + wn * 64 + c * 16 + coln;
      const int mrow = m0 + wm * 32 + rb * 16 + half8;
#pragma unroll
      for (int j = 0; j < 8; ++j)
        store_val(&C[(size_t)(mrow + j) * N + n], acc[rb][c][j]);
    }
}

// ---------------------------------------------------------------------------
__global__ void cvt_f32_bf16(const float* __restrict__ in, __bf16* __restrict__ out, int n) {
  int i = blockIdx.x * 256 + threadIdx.x;
  if (i < n) out[i] = (__bf16)in[i];
}

// RoPE in place on bf16 [ntok, heads*64] (row stride ld). One thread per
// (token, head, pair d<32): writes both d and d+32 -> no cross-thread hazard.
__global__ void rope_kernel(__bf16* __restrict__ t, const float* __restrict__ cosT,
                            const float* __restrict__ sinT, int heads, int ld, int ntok) {
  int idx = blockIdx.x * 256 + threadIdx.x;
  int total = ntok * heads * 32;
  if (idx >= total) return;
  int d   = idx & 31;
  int r   = idx >> 5;
  int h   = r % heads;
  int tok = r / heads;
  __bf16* p = t + (size_t)tok * ld + h * DH;
  float a = (float)p[d];
  float b = (float)p[d + 32];
  float c0 = cosT[tok * DH + d],      s0 = sinT[tok * DH + d];
  float c1 = cosT[tok * DH + d + 32], s1 = sinT[tok * DH + d + 32];
  p[d]      = (__bf16)(a * c0 - b * s0);  // rot_half: -t[d+32] for d<32
  p[d + 32] = (__bf16)(b * c1 + a * s1);  // rot_half: +t[d]   for d>=32
}

// V[n][kvh*64+d] -> Vt[kvh][d][n]  (so P*V B-fragments are contiguous loads)
__global__ void transpose_v_kernel(const __bf16* __restrict__ V, __bf16* __restrict__ Vt,
                                   int ntok) {
  int idx = blockIdx.x * 256 + threadIdx.x;
  if (idx >= KVD * ntok) return;
  int tok  = idx % ntok;
  int rest = idx / ntok;           // kvh*64 + d
  Vt[idx] = V[(size_t)tok * KVD + rest];
}

// ---------------------------------------------------------------------------
// Flash attention: one wave per (head, 16-query-row block). Causal.
//   S(16x32) = Q(16x64) K^T   -> 4 wmma per key block
//   online softmax in C-fragment layout (rows = VGPR j + half*8)
//   P -> bf16 via per-wave LDS tile -> A-fragment; O += P * V -> 4 wmma
// ---------------------------------------------------------------------------
__global__ __launch_bounds__(256) void attn_kernel(const __bf16* __restrict__ Q,
                                                   const __bf16* __restrict__ K,
                                                   const __bf16* __restrict__ Vt,
                                                   __bf16* __restrict__ O) {
  __shared__ alignas(16) __bf16 smem[8 * 16 * PLDS_STRIDE];
  const int wave  = threadIdx.x >> 5;
  const int lane  = threadIdx.x & 31;
  const int task  = blockIdx.x * 8 + wave;
  const int h     = task & (NHQ - 1);   // waves in a block share qblk -> equal trip counts
  const int qblk  = task >> 5;          // 0..127
  const int kvh   = h >> 3;             // REP = 8
  const int coln  = lane & 15;
  const int half8 = (lane >> 4) << 3;
  __bf16* pls = &smem[wave * 16 * PLDS_STRIDE];

  const float scale = 0.125f;  // 1/sqrt(64)

  bfx16 qf0 = load_frag(Q, DMODEL, qblk * 16, h * DH + 0);
  bfx16 qf1 = load_frag(Q, DMODEL, qblk * 16, h * DH + 32);

  fx8 accO[4] = {};
  float m_run[8], l_run[8];
#pragma unroll
  for (int j = 0; j < 8; ++j) { m_run[j] = -1e30f; l_run[j] = 0.f; }

  const int kmax = qblk * 16 + 15;
  for (int s0 = 0; s0 <= kmax; s0 += 32) {
    // ---- S = Q K^T for 32 keys (two 16-key tiles) ----
    float sv[2][8];
#pragma unroll
    for (int t = 0; t < 2; ++t) {
      bfx16 kf0 = load_frag(K, KVD, s0 + t * 16, kvh * DH + 0);
      bfx16 kf1 = load_frag(K, KVD, s0 + t * 16, kvh * DH + 32);
      fx8 s = {};
      s = __builtin_amdgcn_wmma_f32_16x16x32_bf16(false, qf0, false, kf0, (short)0, s, false, false);
      s = __builtin_amdgcn_wmma_f32_16x16x32_bf16(false, qf1, false, kf1, (short)0, s, false, false);
#pragma unroll
      for (int j = 0; j < 8; ++j) {
        int m  = qblk * 16 + half8 + j;
        int sc = s0 + t * 16 + coln;
        float v = s[j] * scale;
        sv[t][j] = (sc <= m) ? v : -1e30f;   // causal mask
      }
    }
    // ---- online softmax per row j (rows align with accO rows) ----
#pragma unroll
    for (int j = 0; j < 8; ++j) {
      float rm = fmaxf(sv[0][j], sv[1][j]);
#pragma unroll
      for (int off = 1; off < 16; off <<= 1)
        rm = fmaxf(rm, __shfl_xor(rm, off));      // stays within 16-lane halves
      float nm    = fmaxf(m_run[j], rm);
      float alpha = __expf(m_run[j] - nm);
      float p0 = __expf(sv[0][j] - nm);
      float p1 = __expf(sv[1][j] - nm);
      sv[0][j] = p0; sv[1][j] = p1;
      float ps = p0 + p1;
#pragma unroll
      for (int off = 1; off < 16; off <<= 1)
        ps += __shfl_xor(ps, off);
      l_run[j] = l_run[j] * alpha + ps;
      m_run[j] = nm;
#pragma unroll
      for (int c = 0; c < 4; ++c) accO[c][j] *= alpha;
    }
    // ---- P (C layout) -> LDS -> A-fragment layout ----
#pragma unroll
    for (int t = 0; t < 2; ++t)
#pragma unroll
      for (int j = 0; j < 8; ++j)
        pls[(half8 + j) * PLDS_STRIDE + t * 16 + coln] = (__bf16)sv[t][j];
    asm volatile("s_wait_dscnt 0" ::: "memory");  // wave-level LDS fence (lanes lockstep)
    bfx16 pf;
    {
      const __bf16* pp = pls + coln * PLDS_STRIDE + half8;
      bfx8 lo = *(const bfx8*)pp;
      bfx8 hi = *(const bfx8*)(pp + 16);
#pragma unroll
      for (int i = 0; i < 8; ++i) { pf[i] = lo[i]; pf[i + 8] = hi[i]; }
    }
    // ---- O += P * V  (V^T rows are d, contiguous in s) ----
#pragma unroll
    for (int c = 0; c < 4; ++c) {
      bfx16 vf = load_frag(Vt + (size_t)kvh * DH * NTOK, NTOK, c * 16, s0);
      accO[c] = __builtin_amdgcn_wmma_f32_16x16x32_bf16(
          false, pf, false, vf, (short)0, accO[c], false, false);
    }
  }
  // ---- normalize and store O[m][h*64+d] ----
#pragma unroll
  for (int c = 0; c < 4; ++c) {
    int d = h * DH + c * 16 + coln;
#pragma unroll
    for (int j = 0; j < 8; ++j) {
      int m = qblk * 16 + half8 + j;
      O[(size_t)m * DMODEL + d] = (__bf16)(accO[c][j] / l_run[j]);
    }
  }
}

// ---------------------------------------------------------------------------
extern "C" void kernel_launch(void* const* d_in, const int* in_sizes, int n_in,
                              void* d_out, int out_size, void* d_ws, size_t ws_size,
                              hipStream_t stream) {
  (void)in_sizes; (void)n_in; (void)out_size; (void)ws_size;
  const float* x    = (const float*)d_in[0];
  const float* Wq   = (const float*)d_in[1];
  const float* Wk   = (const float*)d_in[2];
  const float* Wv   = (const float*)d_in[3];
  const float* Wo   = (const float*)d_in[4];
  const float* cosT = (const float*)d_in[5];
  const float* sinT = (const float*)d_in[6];
  float* out = (float*)d_out;

  // bump allocator on workspace (~29 MB total with reuse)
  char*  base = (char*)d_ws;
  size_t off  = 0;
  auto alloc = [&](size_t bytes) -> void* {
    void* p = base + off;
    off += (bytes + 255) & ~(size_t)255;
    return p;
  };
  __bf16* xb  = (__bf16*)alloc((size_t)NTOK * DMODEL * 2);   // reused as O after V gemm
  __bf16* Wqb = (__bf16*)alloc((size_t)DMODEL * DMODEL * 2); // reused as Wo after Q gemm
  __bf16* Wkb = (__bf16*)alloc((size_t)KVD * DMODEL * 2);
  __bf16* Wvb = (__bf16*)alloc((size_t)KVD * DMODEL * 2);
  __bf16* Qb  = (__bf16*)alloc((size_t)NTOK * DMODEL * 2);
  __bf16* Kb  = (__bf16*)alloc((size_t)NTOK * KVD * 2);
  __bf16* Vb  = (__bf16*)alloc((size_t)NTOK * KVD * 2);
  __bf16* Vtb = (__bf16*)alloc((size_t)KVD * NTOK * 2);
  __bf16* Wob = Wqb;  // reuse (stream-ordered: converted after Q gemm reads Wqb)
  __bf16* Ob  = xb;   // reuse (stream-ordered: written after all x-gemms)

  auto cdiv = [](int a, int b) { return (a + b - 1) / b; };

  // 1) fp32 -> bf16 conversions
  cvt_f32_bf16<<<cdiv(NTOK * DMODEL, 256), 256, 0, stream>>>(x,  xb,  NTOK * DMODEL);
  cvt_f32_bf16<<<cdiv(DMODEL * DMODEL, 256), 256, 0, stream>>>(Wq, Wqb, DMODEL * DMODEL);
  cvt_f32_bf16<<<cdiv(KVD * DMODEL, 256), 256, 0, stream>>>(Wk, Wkb, KVD * DMODEL);
  cvt_f32_bf16<<<cdiv(KVD * DMODEL, 256), 256, 0, stream>>>(Wv, Wvb, KVD * DMODEL);

  // 2) projections: Q = x Wq^T, K = x Wk^T, V = x Wv^T   (128x128 tiles)
  gemm_abt_kernel<__bf16><<<dim3(NTOK / 128, DMODEL / 128), 256, 0, stream>>>(
      xb, Wqb, Qb, NTOK, DMODEL, DMODEL);
  gemm_abt_kernel<__bf16><<<dim3(NTOK / 128, KVD / 128), 256, 0, stream>>>(
      xb, Wkb, Kb, NTOK, KVD, DMODEL);
  gemm_abt_kernel<__bf16><<<dim3(NTOK / 128, KVD / 128), 256, 0, stream>>>(
      xb, Wvb, Vb, NTOK, KVD, DMODEL);

  // 3) Wo conversion (into reused Wq slot), RoPE, V transpose
  cvt_f32_bf16<<<cdiv(DMODEL * DMODEL, 256), 256, 0, stream>>>(Wo, Wob, DMODEL * DMODEL);
  rope_kernel<<<cdiv(NTOK * NHQ * 32, 256), 256, 0, stream>>>(Qb, cosT, sinT, NHQ, DMODEL, NTOK);
  rope_kernel<<<cdiv(NTOK * NHKV * 32, 256), 256, 0, stream>>>(Kb, cosT, sinT, NHKV, KVD, NTOK);
  transpose_v_kernel<<<cdiv(KVD * NTOK, 256), 256, 0, stream>>>(Vb, Vtb, NTOK);

  // 4) causal flash attention -> O (bf16, reuses x slot)
  attn_kernel<<<dim3((NHQ * (NTOK / 16)) / 8), 256, 0, stream>>>(Qb, Kb, Vtb, Ob);

  // 5) out = O Wo^T (fp32 output)
  gemm_abt_kernel<float><<<dim3(NTOK / 128, DMODEL / 128), 256, 0, stream>>>(
      Ob, Wob, out, NTOK, DMODEL, DMODEL);
}
